// rosa_slow_4bit_layer_71992241815995
// MI455X (gfx1250) — 8-bit matrix operand order repaired and checked by reading, compiled, not run on hardware
//
#include <hip/hip_runtime.h>


#define TT 1024
#define CDIM 512
#define GG 128

typedef int v8i __attribute__((ext_vector_type(8)));

__global__ __launch_bounds__(1024)
void rosa_wmma_kernel(const float* __restrict__ q, const float* __restrict__ k,
                      const float* __restrict__ v, const float* __restrict__ emb,
                      float* __restrict__ out) {
  __shared__ int qsym[TT];
  __shared__ int ksym[TT];
  __shared__ unsigned char eq8[16 * TT];
  __shared__ int Lbuf[2][TT + 1];
  __shared__ int lnArr[TT];
  __shared__ int ebArr[TT];
  __shared__ int wmax[2][32];

  const int tid  = threadIdx.x;
  const int lane = tid & 31;
  const int wv   = tid >> 5;
  const int blk  = blockIdx.x;
  const int b    = blk / GG;
  const int g    = blk % GG;

  {
    const float* qrow = q + ((size_t)(b * TT + tid)) * CDIM + g * 4;
    const float* krow = k + ((size_t)(b * TT + tid)) * CDIM + g * 4;
    int qs = 0, ks = 0;
#pragma unroll
    for (int bb = 0; bb < 4; ++bb) {
      qs |= ((int)(qrow[bb] > 0.0f)) << bb;
      ks |= ((int)(krow[bb] > 0.0f)) << bb;
    }
    qsym[tid] = qs;
    ksym[tid] = ks;
    Lbuf[0][tid] = 0;
    Lbuf[1][tid] = 0;
  }
  __syncthreads();

  for (int i0 = 0; i0 < TT; i0 += 16) {
    const int arow = i0 + (lane & 15);
    const int qa   = qsym[arow];
    const int kb0  = (lane < 16) ? 0 : 8;
    const unsigned da0 = (unsigned)(qa - kb0);
    const unsigned a0  = (da0 < 4u) ? (1u << (da0 * 8u)) : 0u;
    const unsigned da1 = da0 - 4u;
    const unsigned a1  = (da1 < 4u) ? (1u << (da1 * 8u)) : 0u;
    v8i A = { (int)a0, (int)a1, 0, 0, 0, 0, 0, 0 };

#pragma unroll
    for (int t2 = 0; t2 < 2; ++t2) {
      const int e0  = (wv * 2 + t2) * 16;
      const int col = e0 + (lane & 15);
      const int ksy = ksym[col];
      const unsigned bit = 1u << ((unsigned)(ksy & 3) * 8u);
      const int kq = ksy >> 2;
      const bool lo = (lane < 16);
      const unsigned b0 = (kq == (lo ? 0 : 2)) ? bit : 0u;
      const unsigned b1 = (kq == (lo ? 1 : 3)) ? bit : 0u;
      const unsigned b2 = 0u;
      const unsigned b3 = 0u;
      v8i Bv = { (int)b0, (int)b1, (int)b2, (int)b3, 0, 0, 0, 0 };
      v8i Cz = {};
      v8i acc = __builtin_amdgcn_wmma_i32_16x16x64_iu8(false, A, false, Bv, Cz,
                                                       false, false);
      const int rB = (lane < 16) ? 0 : 8;
#pragma unroll
      for (int r = 0; r < 8; ++r)
        eq8[(r + rB) * TT + col] = (unsigned char)acc[r];
    }
    __syncthreads();

    for (int r = 0; r < 16; ++r) {
      const int i = i0 + r;
      const int p = i & 1;
      const int nb  = Lbuf[p][tid];
      const int eqv = (int)eq8[r * TT + tid];
      const int L   = eqv ? (nb + 1) : 0;
      Lbuf[p ^ 1][tid + 1] = L;
      const int m  = (tid < i) ? L : 0;
      int sc = (m << 10) | tid;
#pragma unroll
      for (int off = 16; off >= 1; off >>= 1) {
        const int o = __shfl_xor(sc, off, 32);
        sc = (o > sc) ? o : sc;
      }
      if (lane == 0) wmax[p][wv] = sc;
      __syncthreads();
      if (wv == 0) {
        int s2 = wmax[p][lane];
#pragma unroll
        for (int off = 16; off >= 1; off >>= 1) {
          const int o = __shfl_xor(s2, off, 32);
          s2 = (o > s2) ? o : s2;
        }
        if (lane == 0) { lnArr[i] = s2 >> 10; ebArr[i] = s2 & (TT - 1); }
      }
    }
    __syncthreads();
  }

  {
    const int ln = lnArr[tid];
    int s = ebArr[tid] + 1;
    if (s > TT - 1) s = TT - 1;
    const float* vrow = v + ((size_t)(b * TT + s)) * CDIM + g * 4;
    float* orow = out + ((size_t)(b * TT + tid)) * CDIM + g * 4;
#pragma unroll
    for (int bb = 0; bb < 4; ++bb) {
      float val = 0.0f;
      if (ln > 0) {
        const float sgn = (vrow[bb] > 0.0f) ? 1.0f : -1.0f;
        val = sgn * emb[g * 4 + bb];
      }
      orow[bb] = val;
    }
  }
}

extern "C" void kernel_launch(void* const* d_in, const int* in_sizes, int n_in,
                              void* d_out, int out_size, void* d_ws, size_t ws_size,
                              hipStream_t stream) {
  (void)in_sizes; (void)n_in; (void)out_size; (void)d_ws; (void)ws_size;
  const float* q   = (const float*)d_in[0];
  const float* k   = (const float*)d_in[1];
  const float* v   = (const float*)d_in[2];
  const float* emb = (const float*)d_in[3];
  float* out = (float*)d_out;
  rosa_wmma_kernel<<<dim3(4 * GG), dim3(1024), 0, stream>>>(q, k, v, emb, out);
}
